// MultiHeadAttetion_22067541967602
// MI455X (gfx1250) — compile-verified, matrix operand order checked by reading, not run on hardware
//
#include <hip/hip_runtime.h>
#include <hip/hip_bf16.h>

// ---------------------------------------------------------------------------
// MHA for B=4, S=2048, D=1024, H=16, HD=64 on gfx1250 (MI455X).
// bf16 WMMA (v_wmma_f32_16x16x32_bf16), f32 accumulate.
// Tile staging uses CDNA5 async global->LDS copies (GLOBAL_LOAD_ASYNC_TO_LDS,
// ASYNCcnt) so the copy engine overlaps the WMMA pipe with no VGPR staging.
// cvt(f32->bf16) -> GEMM(Q) -> GEMM(K) -> GEMM(V^T) -> flash attn -> GEMM(O).
// ---------------------------------------------------------------------------

typedef __attribute__((ext_vector_type(16))) __bf16 bf16x16;
typedef __attribute__((ext_vector_type(8)))  __bf16 bf16x8;
typedef __attribute__((ext_vector_type(8)))  float  f32x8;

union FragU { bf16x16 v; bf16x8 h[2]; };

static constexpr int Bc = 4, Sc = 2048, Dc = 1024, Hc = 16, HDc = 64;
static constexpr int Mrows = Bc * Sc; // 8192

// Async 16B global->LDS copy, tracked on ASYNCcnt (no VGPR data staging).
// VDST VGPR = LDS byte address (low 32 bits of a generic pointer to __shared__).
__device__ __forceinline__ void async_copy_b128(void* lds, const void* gptr) {
  unsigned la = (unsigned)(size_t)lds;
  asm volatile("global_load_async_to_lds_b128 %0, %1, off"
               :: "v"(la), "v"(gptr) : "memory");
}
__device__ __forceinline__ void wait_async0() {
  asm volatile("s_wait_asynccnt 0" ::: "memory");
}

// ---------------------------------------------------------------------------
__global__ __launch_bounds__(256)
void cvt_f32_to_bf16(const float* __restrict__ in, unsigned short* __restrict__ outp, int n) {
  __bf16* out = reinterpret_cast<__bf16*>(outp);
  int i = (blockIdx.x * blockDim.x + threadIdx.x) * 4;
  if (i + 3 < n) {
    float4 v = *(const float4*)(in + i);
    out[i + 0] = (__bf16)v.x;
    out[i + 1] = (__bf16)v.y;
    out[i + 2] = (__bf16)v.z;
    out[i + 3] = (__bf16)v.w;
  }
}

// A/B fragment (16x32 bf16): lane l holds row l%16; K chunks {0..7,16..23}
// (lanes 0-15) / {8..15,24..31} (lanes 16-31) -> two contiguous 16B loads.
__device__ __forceinline__ bf16x16 load_frag_row(const __bf16* rowptr, int half) {
  FragU u;
  u.h[0] = *(const bf16x8*)(rowptr + half * 8);
  u.h[1] = *(const bf16x8*)(rowptr + 16 + half * 8);
  return u.v;
}

// ---------------------------------------------------------------------------
// GEMM: out[M,N] = A[M,K] bf16 @ W[N,K]^T bf16 + bias[N]
// Block tile 128(M) x 256(N) x 32(K); 8 waves as 2(M) x 4(N); wave tile 64x64
// = 4x4 WMMA accums -> 16 WMMAs per 8 LDS fragment loads per K-step.
// Async-copy double-buffered LDS, 1 barrier / K-step.
// MODE 0: bf16 rm; MODE 1: V^T bf16 [B,H,HD,S]; MODE 2: f32 rm.
// ---------------------------------------------------------------------------
template <int MODE>
__global__ __launch_bounds__(256)
void gemm_bf16_wmma(const unsigned short* __restrict__ Ap,
                    const unsigned short* __restrict__ Wp,
                    const float* __restrict__ bias,
                    void* __restrict__ out,
                    int M, int N, int K) {
  const __bf16* A = reinterpret_cast<const __bf16*>(Ap);
  const __bf16* W = reinterpret_cast<const __bf16*>(Wp);

  constexpr int BM = 128, BN = 256, BK = 32;
  __shared__ __bf16 As[2][BM][BK]; // 16 KB
  __shared__ __bf16 Ws[2][BN][BK]; // 32 KB

  const int tid  = threadIdx.x;
  const int lane = tid & 31;
  const int wid  = tid >> 5;
  const int half = lane >> 4;
  const int l16  = lane & 15;
  const int wm   = wid & 1;   // 2 waves over M
  const int wn   = wid >> 1;  // 4 waves over N
  const int tileM = blockIdx.y * BM;
  const int tileN = blockIdx.x * BN;

  // Per-thread global->LDS chunk coordinates (8 bf16 per 16B chunk).
  int arow[2], acol[2], wrow[4], wcol[4];
  #pragma unroll
  for (int i = 0; i < 2; ++i) { int ch = tid + i * 256; arow[i] = ch >> 2; acol[i] = (ch & 3) * 8; }
  #pragma unroll
  for (int i = 0; i < 4; ++i) { int ch = tid + i * 256; wrow[i] = ch >> 2; wcol[i] = (ch & 3) * 8; }

  f32x8 acc[4][4];
  #pragma unroll
  for (int mi = 0; mi < 4; ++mi)
    #pragma unroll
    for (int ni = 0; ni < 4; ++ni)
      acc[mi][ni] = (f32x8)0.0f;

  float bv[4];
  #pragma unroll
  for (int ni = 0; ni < 4; ++ni)
    bv[ni] = bias[tileN + wn * 64 + ni * 16 + l16];

  // Issue async copies of K-slice k0 into LDS buffer `buf`.
  auto aload = [&](int buf, int k0) {
    #pragma unroll
    for (int i = 0; i < 2; ++i)
      async_copy_b128(&As[buf][arow[i]][acol[i]],
                      &A[(size_t)(tileM + arow[i]) * K + k0 + acol[i]]);
    #pragma unroll
    for (int i = 0; i < 4; ++i)
      async_copy_b128(&Ws[buf][wrow[i]][wcol[i]],
                      &W[(size_t)(tileN + wrow[i]) * K + k0 + wcol[i]]);
  };

  const int KT = K / BK;
  aload(0, 0);
  wait_async0();
  __syncthreads();

  for (int kt = 0; kt < KT; ++kt) {
    const int cur = kt & 1;
    if (kt + 1 < KT) aload(1 - cur, (kt + 1) * BK); // async, overlaps WMMAs

    bf16x16 af[4], wf[4];
    #pragma unroll
    for (int mi = 0; mi < 4; ++mi)
      af[mi] = load_frag_row(&As[cur][wm * 64 + mi * 16 + l16][0], half);
    #pragma unroll
    for (int ni = 0; ni < 4; ++ni)
      wf[ni] = load_frag_row(&Ws[cur][wn * 64 + ni * 16 + l16][0], half);

    #pragma unroll
    for (int mi = 0; mi < 4; ++mi)
      #pragma unroll
      for (int ni = 0; ni < 4; ++ni)
        acc[mi][ni] = __builtin_amdgcn_wmma_f32_16x16x32_bf16(
            false, af[mi], false, wf[ni], (short)0, acc[mi][ni], false, false);

    wait_async0();
    __syncthreads();
  }

  // C/D layout: VGPR r, lane l -> M = r + 8*(l>=16), N = l%16.
  #pragma unroll
  for (int mi = 0; mi < 4; ++mi) {
    #pragma unroll
    for (int ni = 0; ni < 4; ++ni) {
      const f32x8 c = acc[mi][ni];
      #pragma unroll
      for (int r = 0; r < 8; ++r) {
        int m = tileM + wm * 64 + mi * 16 + r + 8 * half;
        int n = tileN + wn * 64 + ni * 16 + l16;
        float v = c[r] + bv[ni];
        if (MODE == 0) {
          ((__bf16*)out)[(size_t)m * N + n] = (__bf16)v;
        } else if (MODE == 2) {
          ((float*)out)[(size_t)m * N + n] = v;
        } else { // V^T -> [B, H, HD, S]
          int bb = m >> 11, ss = m & (Sc - 1);
          int hh = n >> 6,  dd = n & (HDc - 1);
          ((__bf16*)out)[((size_t)((bb * Hc + hh) * HDc + dd)) * Sc + ss] = (__bf16)v;
        }
      }
    }
  }
}

// Cross-lane reductions within each 16-lane half (wave32).
__device__ __forceinline__ float rowmax16(float v) {
  #pragma unroll
  for (int m = 1; m < 16; m <<= 1) v = fmaxf(v, __shfl_xor(v, m, 32));
  return v;
}
__device__ __forceinline__ float rowsum16(float v) {
  #pragma unroll
  for (int m = 1; m < 16; m <<= 1) v += __shfl_xor(v, m, 32);
  return v;
}

// ---------------------------------------------------------------------------
// Flash attention, causal, log2-domain softmax.
// 256 threads = 8 waves; block handles 128 queries of one (b,h); each wave
// owns a 16-query tile. 32-key K/V tiles staged cooperatively with async
// global->LDS copies (double-buffered, 1 barrier per tile). P transposed
// through per-wave LDS with an explicit s_wait_dscnt 0.
// ---------------------------------------------------------------------------
__global__ __launch_bounds__(256)
void flash_attn_wmma(const unsigned short* __restrict__ Qp,
                     const unsigned short* __restrict__ Kp,
                     const unsigned short* __restrict__ Vp,
                     unsigned short* __restrict__ Op) {
  const __bf16* Qb = reinterpret_cast<const __bf16*>(Qp);
  const __bf16* Kb = reinterpret_cast<const __bf16*>(Kp);
  const __bf16* Vt = reinterpret_cast<const __bf16*>(Vp);
  __bf16* Ob = reinterpret_cast<__bf16*>(Op);

  // softmax in log2 domain: exp2 is native v_exp_f32; fold 1/sqrt(64)*log2(e).
  const float scale2 = 0.125f * 1.44269504088896340736f;

  const int tid  = threadIdx.x;
  const int lane = tid & 31;
  const int w    = tid >> 5;       // wave id 0..7
  const int half = lane >> 4;
  const int l16  = lane & 15;
  const int b = blockIdx.z, h = blockIdx.y;
  const int qblock = blockIdx.x * 128;
  const int q0 = qblock + w * 16;

  __shared__ __bf16 Ks[2][32][64];    // 8 KB  (keys x dims)
  __shared__ __bf16 Vs[2][64][32];    // 8 KB  (dims x keys)
  __shared__ __bf16 plds[8][16][32];  // 8 KB  per-wave P transpose

  // Resident Q fragments (16 rows x HD=64 = two 16x32 A-frags).
  const __bf16* qbase = Qb + ((size_t)(b * Sc + q0 + l16)) * Dc + h * HDc;
  bf16x16 qf[2];
  #pragma unroll
  for (int dc = 0; dc < 2; ++dc)
    qf[dc] = load_frag_row(qbase + dc * 32, half);

  f32x8 o[4];
  #pragma unroll
  for (int dt = 0; dt < 4; ++dt) o[dt] = (f32x8)0.0f;
  float mi[8], li[8];
  #pragma unroll
  for (int r = 0; r < 8; ++r) { mi[r] = -1e30f; li[r] = 0.0f; }

  // Per-thread cooperative tile chunk coords (one 16B chunk each for K and V).
  const int krow = tid >> 3, kcol = (tid & 7) * 8;   // Ks: 32x64
  const int vrow = tid >> 2, vcol = (tid & 3) * 8;   // Vs: 64x32
  const __bf16* kg = Kb + ((size_t)(b * Sc + krow)) * Dc + h * HDc + kcol;
  const __bf16* vg = Vt + ((size_t)((b * Hc + h) * HDc + vrow)) * Sc + vcol;

  auto aload = [&](int buf, int kt) {
    const int k0 = kt * 32;
    async_copy_b128(&Ks[buf][krow][kcol], kg + (size_t)k0 * Dc);
    async_copy_b128(&Vs[buf][vrow][vcol], vg + k0);
  };

  const int nkt = (qblock + 128) >> 5;   // block-level causal bound
  aload(0, 0);
  wait_async0();
  __syncthreads();

  for (int kt = 0; kt < nkt; ++kt) {
    const int cur = kt & 1;
    const int k0  = kt * 32;
    if (kt + 1 < nkt) aload(1 - cur, kt + 1);  // async, overlaps compute

    // Tile fully above this wave's diagonal -> contributes nothing.
    if (k0 <= q0 + 15) {
      // S = Q @ K^T
      f32x8 s[2];
      s[0] = (f32x8)0.0f; s[1] = (f32x8)0.0f;
      #pragma unroll
      for (int sub = 0; sub < 2; ++sub)
        #pragma unroll
        for (int dc = 0; dc < 2; ++dc) {
          bf16x16 kf = load_frag_row(&Ks[cur][sub * 16 + l16][dc * 32], half);
          s[sub] = __builtin_amdgcn_wmma_f32_16x16x32_bf16(
              false, qf[dc], false, kf, (short)0, s[sub], false, false);
        }

      const bool needMask = (k0 + 31 > q0);  // tile straddles the diagonal
      #pragma unroll
      for (int r = 0; r < 8; ++r) {
        const int qg = q0 + r + 8 * half;
        float s0 = s[0][r] * scale2;
        float s1 = s[1][r] * scale2;
        if (needMask) {
          if (k0 + l16      > qg) s0 = -1e30f;
          if (k0 + 16 + l16 > qg) s1 = -1e30f;
        }
        float mnew = fmaxf(mi[r], rowmax16(fmaxf(s0, s1)));
        float al = exp2f(mi[r] - mnew);
        float p0 = exp2f(s0 - mnew);
        float p1 = exp2f(s1 - mnew);
        li[r] = li[r] * al + rowsum16(p0 + p1);
        mi[r] = mnew;
        #pragma unroll
        for (int dt = 0; dt < 4; ++dt) o[dt][r] *= al;
        const int row = r + 8 * half;
        plds[w][row][l16]      = (__bf16)p0;
        plds[w][row][16 + l16] = (__bf16)p1;
      }
      asm volatile("s_wait_dscnt 0" ::: "memory"); // per-wave LDS roundtrip

      bf16x16 pf = load_frag_row(&plds[w][l16][0], half);
      #pragma unroll
      for (int dt = 0; dt < 4; ++dt) {
        bf16x16 vf = load_frag_row(&Vs[cur][dt * 16 + l16][0], half);
        o[dt] = __builtin_amdgcn_wmma_f32_16x16x32_bf16(
            false, pf, false, vf, (short)0, o[dt], false, false);
      }
    }

    wait_async0();
    __syncthreads();
  }

  #pragma unroll
  for (int r = 0; r < 8; ++r) {
    const float inv = 1.0f / li[r];
    const int q = q0 + r + 8 * half;
    #pragma unroll
    for (int dt = 0; dt < 4; ++dt) {
      float v = o[dt][r] * inv;
      Ob[((size_t)(b * Sc + q)) * Dc + h * HDc + dt * 16 + l16] = (__bf16)v;
    }
  }
}

// ---------------------------------------------------------------------------
extern "C" void kernel_launch(void* const* d_in, const int* in_sizes, int n_in,
                              void* d_out, int out_size, void* d_ws, size_t ws_size,
                              hipStream_t stream) {
  (void)in_sizes; (void)n_in; (void)out_size; (void)ws_size;
  const float* X  = (const float*)d_in[0];
  const float* Wq = (const float*)d_in[1];
  const float* bq = (const float*)d_in[2];
  const float* Wk = (const float*)d_in[3];
  const float* bk = (const float*)d_in[4];
  const float* Wv = (const float*)d_in[5];
  const float* bv = (const float*)d_in[6];
  const float* Wo = (const float*)d_in[7];
  const float* bo = (const float*)d_in[8];

  char* wsp = (char*)d_ws;
  auto carve = [&](size_t bytes) -> unsigned short* {
    unsigned short* p = (unsigned short*)wsp;
    wsp += (bytes + 255) & ~(size_t)255;
    return p;
  };
  const size_t nX = (size_t)Mrows * Dc;
  const size_t nW = (size_t)Dc * Dc;
  unsigned short* Xb  = carve(nX * 2);
  unsigned short* Wqb = carve(nW * 2);
  unsigned short* Wkb = carve(nW * 2);
  unsigned short* Wvb = carve(nW * 2);
  unsigned short* Wob = carve(nW * 2);
  unsigned short* Qb  = carve(nX * 2);
  unsigned short* Kb  = carve(nX * 2);
  unsigned short* Vt  = carve(nX * 2);
  unsigned short* Ab  = carve(nX * 2);

  cvt_f32_to_bf16<<<(int)(nX / 1024), 256, 0, stream>>>(X,  Xb,  (int)nX);
  cvt_f32_to_bf16<<<(int)(nW / 1024), 256, 0, stream>>>(Wq, Wqb, (int)nW);
  cvt_f32_to_bf16<<<(int)(nW / 1024), 256, 0, stream>>>(Wk, Wkb, (int)nW);
  cvt_f32_to_bf16<<<(int)(nW / 1024), 256, 0, stream>>>(Wv, Wvb, (int)nW);
  cvt_f32_to_bf16<<<(int)(nW / 1024), 256, 0, stream>>>(Wo, Wob, (int)nW);

  dim3 gg(Dc / 256, Mrows / 128);
  dim3 gb(256);
  gemm_bf16_wmma<0><<<gg, gb, 0, stream>>>(Xb, Wqb, bq, (void*)Qb, Mrows, Dc, Dc);
  gemm_bf16_wmma<0><<<gg, gb, 0, stream>>>(Xb, Wkb, bk, (void*)Kb, Mrows, Dc, Dc);
  gemm_bf16_wmma<1><<<gg, gb, 0, stream>>>(Xb, Wvb, bv, (void*)Vt, Mrows, Dc, Dc);

  dim3 ga(Sc / 128, Hc, Bc);
  flash_attn_wmma<<<ga, dim3(256), 0, stream>>>(Qb, Kb, Vt, Ab);

  gemm_bf16_wmma<2><<<gg, gb, 0, stream>>>(Ab, Wob, bo, d_out, Mrows, Dc, Dc);
}
